// MultiHeadAttention_51178830299562
// MI455X (gfx1250) — matrix operand order repaired and checked by reading, not compiled, not run on hardware
//
#include <hip/hip_runtime.h>
#include <cstdint>

#define D_MODEL 1024
#define SEQ     4096
#define NUM_HEADS 16
#define D_K     64

typedef __bf16 v16bf __attribute__((ext_vector_type(16)));
typedef float  v8f   __attribute__((ext_vector_type(8)));

union FragB16 { v16bf v; uint4 q[2]; };

__device__ __forceinline__ unsigned short f2bf(float f) {
    unsigned u = __float_as_uint(f);
    u += 0x7FFFu + ((u >> 16) & 1u);          // round-to-nearest-even
    return (unsigned short)(u >> 16);
}

__device__ __forceinline__ unsigned lds_off(const void* p) {
    return (unsigned)(uintptr_t)p;            // LDS aperture: addr[31:0] = LDS offset
}

__device__ __forceinline__ void async_b128(unsigned lds, const void* g) {
    asm volatile("global_load_async_to_lds_b128 %0, %1, off"
                 :: "v"(lds), "v"((unsigned long long)(uintptr_t)g) : "memory");
}

// ---------------------------------------------------------------- conversions

__global__ void cvt_f32_to_bf16(const float* __restrict__ in,
                                unsigned short* __restrict__ out, int n4) {
    int i = blockIdx.x * blockDim.x + threadIdx.x;
    if (i >= n4) return;
    float4 f = reinterpret_cast<const float4*>(in)[i];
    union { uint2 u; unsigned short s[4]; } pk;
    pk.s[0] = f2bf(f.x); pk.s[1] = f2bf(f.y);
    pk.s[2] = f2bf(f.z); pk.s[3] = f2bf(f.w);
    reinterpret_cast<uint2*>(out)[i] = pk.u;
}

// Wt[n][k] = bf16(W[k][n]); rows = K-extent of W, cols = N-extent of W
__global__ void transpose_cvt_bf16(const float* __restrict__ W,
                                   unsigned short* __restrict__ Wt,
                                   int rows, int cols) {
    __shared__ float tile[32][33];
    int c0 = blockIdx.x * 32, r0 = blockIdx.y * 32;
    int tx = threadIdx.x, ty = threadIdx.y;   // block (32, 8)
#pragma unroll
    for (int i = 0; i < 32; i += 8)
        tile[ty + i][tx] = W[(size_t)(r0 + ty + i) * cols + c0 + tx];
    __syncthreads();
#pragma unroll
    for (int i = 0; i < 32; i += 8)
        Wt[(size_t)(c0 + ty + i) * rows + r0 + tx] = f2bf(tile[tx][ty + i]);
}

// ---------------------------------------------------------------- bf16 GEMM
// C[M,N] = A[M,K] (bf16, row-major) * Bt[N,K] (bf16, pre-transposed) + bias[N]
// OUT_MODE 0: bf16 row-major; 1: bf16 transposed [N,M]; 2: fp32 row-major.
// Block = 128 threads (4 waves); wave covers 32(M) x 64(N): 8 wmma / 12 loads.
// Grid (M/32, N/256).

template <int OUT_MODE>
__global__ __launch_bounds__(128)
void gemm_bf16(const unsigned short* __restrict__ A,
               const unsigned short* __restrict__ Bt,
               const float* __restrict__ bias,
               void* __restrict__ Cout, int M, int N, int K) {
    const int lane = threadIdx.x & 31;
    const int wave = threadIdx.x >> 5;
    const int lm   = lane & 15;
    const int half = lane >> 4;
    const int m0   = blockIdx.x * 32;
    const int n0   = blockIdx.y * 256 + wave * 64;

    const unsigned short* ar0 = A + (size_t)(m0 + lm) * K;
    const unsigned short* ar1 = A + (size_t)(m0 + 16 + lm) * K;
    const unsigned short* b0r = Bt + (size_t)(n0 +  0 + lm) * K;
    const unsigned short* b1r = Bt + (size_t)(n0 + 16 + lm) * K;
    const unsigned short* b2r = Bt + (size_t)(n0 + 32 + lm) * K;
    const unsigned short* b3r = Bt + (size_t)(n0 + 48 + lm) * K;
    const int aoff = half * 16;   // A lane layout: K 0..15 / 16..31
    const int boff = half * 16;   // B lane layout: K 0..15 / 16..31

    v8f acc[2][4] = {};
    for (int kb = 0; kb < K; kb += 32) {
        __builtin_prefetch(ar0 + kb + 256, 0, 1);
        __builtin_prefetch(b0r + kb + 256, 0, 1);
        FragB16 a0, a1, b0, b1, b2, b3;
        a0.q[0] = *(const uint4*)(ar0 + kb + aoff);
        a0.q[1] = *(const uint4*)(ar0 + kb + aoff + 8);
        a1.q[0] = *(const uint4*)(ar1 + kb + aoff);
        a1.q[1] = *(const uint4*)(ar1 + kb + aoff + 8);
        b0.q[0] = *(const uint4*)(b0r + kb + boff);
        b0.q[1] = *(const uint4*)(b0r + kb + boff + 8);
        b1.q[0] = *(const uint4*)(b1r + kb + boff);
        b1.q[1] = *(const uint4*)(b1r + kb + boff + 8);
        b2.q[0] = *(const uint4*)(b2r + kb + boff);
        b2.q[1] = *(const uint4*)(b2r + kb + boff + 8);
        b3.q[0] = *(const uint4*)(b3r + kb + boff);
        b3.q[1] = *(const uint4*)(b3r + kb + boff + 8);
        acc[0][0] = __builtin_amdgcn_wmma_f32_16x16x32_bf16(false, a0.v, false, b0.v,
                                                            (short)0, acc[0][0], false, false);
        acc[0][1] = __builtin_amdgcn_wmma_f32_16x16x32_bf16(false, a0.v, false, b1.v,
                                                            (short)0, acc[0][1], false, false);
        acc[0][2] = __builtin_amdgcn_wmma_f32_16x16x32_bf16(false, a0.v, false, b2.v,
                                                            (short)0, acc[0][2], false, false);
        acc[0][3] = __builtin_amdgcn_wmma_f32_16x16x32_bf16(false, a0.v, false, b3.v,
                                                            (short)0, acc[0][3], false, false);
        acc[1][0] = __builtin_amdgcn_wmma_f32_16x16x32_bf16(false, a1.v, false, b0.v,
                                                            (short)0, acc[1][0], false, false);
        acc[1][1] = __builtin_amdgcn_wmma_f32_16x16x32_bf16(false, a1.v, false, b1.v,
                                                            (short)0, acc[1][1], false, false);
        acc[1][2] = __builtin_amdgcn_wmma_f32_16x16x32_bf16(false, a1.v, false, b2.v,
                                                            (short)0, acc[1][2], false, false);
        acc[1][3] = __builtin_amdgcn_wmma_f32_16x16x32_bf16(false, a1.v, false, b3.v,
                                                            (short)0, acc[1][3], false, false);
    }

#pragma unroll
    for (int mt = 0; mt < 2; ++mt) {
#pragma unroll
        for (int j = 0; j < 4; ++j) {
            const int col = n0 + j * 16 + lm;
            const float bv = bias[col];
            if (OUT_MODE == 0) {
                unsigned short* C = (unsigned short*)Cout;
#pragma unroll
                for (int r = 0; r < 8; ++r) {
                    int row = m0 + mt * 16 + half * 8 + r;
                    C[(size_t)row * N + col] = f2bf(acc[mt][j][r] + bv);
                }
            } else if (OUT_MODE == 1) {
                unsigned short* CT = (unsigned short*)Cout;   // [N, M]
                union { uint4 q; unsigned short s[8]; } pk;
#pragma unroll
                for (int r = 0; r < 8; ++r) pk.s[r] = f2bf(acc[mt][j][r] + bv);
                *(uint4*)(CT + (size_t)col * M + m0 + mt * 16 + half * 8) = pk.q;
            } else {
                float* C = (float*)Cout;
#pragma unroll
                for (int r = 0; r < 8; ++r) {
                    int row = m0 + mt * 16 + half * 8 + r;
                    C[(size_t)row * N + col] = acc[mt][j][r] + bv;
                }
            }
        }
    }
}

// ---------------------------------------------------------------- attention
// Q,K bf16 [SEQ, D_MODEL]; Vt bf16 [D_MODEL, SEQ]; Ctx bf16 [SEQ, D_MODEL].
// Grid (SEQ/64, NUM_HEADS), block 128 (4 waves). The 4 waves share each
// 32-key K/V tile through double-buffered LDS filled with async copies
// (GLOBAL_LOAD_ASYNC_TO_LDS_B128, ASYNCcnt), overlapped with WMMA compute.

__global__ __launch_bounds__(128)
void attn_fwd(const unsigned short* __restrict__ Q,
              const unsigned short* __restrict__ Kmat,
              const unsigned short* __restrict__ Vt,
              unsigned short* __restrict__ Ctx) {
    __shared__ unsigned short kbuf[2][32][64];   // keys x d     (4 KB per buffer)
    __shared__ unsigned short vbuf[2][64][32];   // d    x keys  (4 KB per buffer)
    __shared__ unsigned short pbuf[4][16][40];   // per-wave P transpose staging

    const int tid  = threadIdx.x;
    const int lane = tid & 31;
    const int wave = tid >> 5;
    const int lm   = lane & 15;
    const int half = lane >> 4;
    const int head = blockIdx.y;
    const int q0   = blockIdx.x * 64 + wave * 16;
    const float scale = 0.125f;                  // 1/sqrt(64)

    // async-stage the shared K/V tile for key block kb into buffer `buf`
    auto stage_kv = [&](int kb, int buf) {
#pragma unroll
        for (int j = 0; j < 2; ++j) {            // K tile: 32 rows x 128B = 256 chunks
            int c = tid + 128 * j;
            const unsigned short* g =
                Kmat + (size_t)(kb + (c >> 3)) * D_MODEL + head * D_K + (c & 7) * 8;
            async_b128(lds_off(&kbuf[buf][c >> 3][(c & 7) * 8]), g);
        }
#pragma unroll
        for (int j = 0; j < 2; ++j) {            // V tile: 64 rows x 64B = 256 chunks
            int c = tid + 128 * j;
            const unsigned short* g =
                Vt + (size_t)(head * D_K + (c >> 2)) * SEQ + kb + (c & 3) * 8;
            async_b128(lds_off(&vbuf[buf][c >> 2][(c & 3) * 8]), g);
        }
    };

    // Q as two A-fragments (d 0..31, 32..63), loaded once per wave
    const unsigned short* qrow = Q + (size_t)(q0 + lm) * D_MODEL + head * D_K;
    FragB16 a0, a1;
    a0.q[0] = *(const uint4*)(qrow + half * 16);
    a0.q[1] = *(const uint4*)(qrow + half * 16 + 8);
    a1.q[0] = *(const uint4*)(qrow + 32 + half * 16);
    a1.q[1] = *(const uint4*)(qrow + 32 + half * 16 + 8);

    float m[8], l[8];
#pragma unroll
    for (int r = 0; r < 8; ++r) { m[r] = -1e30f; l[r] = 0.0f; }
    v8f o0 = {}, o1 = {}, o2 = {}, o3 = {};

    stage_kv(0, 0);
    asm volatile("s_wait_asynccnt 0" ::: "memory");
    __syncthreads();

    for (int kb = 0; kb < SEQ; kb += 32) {
        const int cur = (kb >> 5) & 1;
        if (kb + 32 < SEQ) stage_kv(kb + 32, cur ^ 1);   // prefetch next tile

        // ---- S = Q K^T for 32 keys (two 16-key column groups), K from LDS
        const unsigned short* kr0 = &kbuf[cur][lm][half * 16];
        const unsigned short* kr1 = &kbuf[cur][16 + lm][half * 16];
        FragB16 k00, k01, k10, k11;
        k00.q[0] = *(const uint4*)(kr0);      k00.q[1] = *(const uint4*)(kr0 + 8);
        k01.q[0] = *(const uint4*)(kr0 + 32); k01.q[1] = *(const uint4*)(kr0 + 40);
        k10.q[0] = *(const uint4*)(kr1);      k10.q[1] = *(const uint4*)(kr1 + 8);
        k11.q[0] = *(const uint4*)(kr1 + 32); k11.q[1] = *(const uint4*)(kr1 + 40);

        v8f c0 = {}, c1 = {};
        c0 = __builtin_amdgcn_wmma_f32_16x16x32_bf16(false, a0.v, false, k00.v,
                                                     (short)0, c0, false, false);
        c0 = __builtin_amdgcn_wmma_f32_16x16x32_bf16(false, a1.v, false, k01.v,
                                                     (short)0, c0, false, false);
        c1 = __builtin_amdgcn_wmma_f32_16x16x32_bf16(false, a0.v, false, k10.v,
                                                     (short)0, c1, false, false);
        c1 = __builtin_amdgcn_wmma_f32_16x16x32_bf16(false, a1.v, false, k11.v,
                                                     (short)0, c1, false, false);

        // ---- online softmax (per-row stats; rows live per lane-half)
#pragma unroll
        for (int r = 0; r < 8; ++r) {
            float s0 = c0[r] * scale, s1 = c1[r] * scale;
            float t = fmaxf(s0, s1);
            t = fmaxf(t, __shfl_xor(t, 1));
            t = fmaxf(t, __shfl_xor(t, 2));
            t = fmaxf(t, __shfl_xor(t, 4));
            t = fmaxf(t, __shfl_xor(t, 8));
            float mn = fmaxf(m[r], t);
            float alpha = __expf(m[r] - mn);
            m[r] = mn;
            float p0 = __expf(s0 - mn);
            float p1 = __expf(s1 - mn);
            c0[r] = p0; c1[r] = p1;
            float s = p0 + p1;
            s += __shfl_xor(s, 1);
            s += __shfl_xor(s, 2);
            s += __shfl_xor(s, 4);
            s += __shfl_xor(s, 8);
            l[r] = l[r] * alpha + s;
            o0[r] *= alpha; o1[r] *= alpha; o2[r] *= alpha; o3[r] *= alpha;
        }

        // ---- transpose P (lane=key) -> A-fragment (lane=query) through LDS
#pragma unroll
        for (int r = 0; r < 8; ++r) {
            int row = half * 8 + r;
            pbuf[wave][row][lm]      = f2bf(c0[r]);
            pbuf[wave][row][lm + 16] = f2bf(c1[r]);
        }
        asm volatile("s_wait_dscnt 0" ::: "memory");
        FragB16 pa;
        {
            const unsigned short* prow = &pbuf[wave][lm][0];
            pa.q[0] = *(const uint4*)(prow + half * 16);
            pa.q[1] = *(const uint4*)(prow + half * 16 + 8);
        }

        // ---- O += P * V, V fragments from LDS (d-major rows, keys contiguous)
        FragB16 v0, v1, v2, v3;
        {
            const unsigned short* vb0 = &vbuf[cur][ 0 + lm][half * 16];
            const unsigned short* vb1 = &vbuf[cur][16 + lm][half * 16];
            const unsigned short* vb2 = &vbuf[cur][32 + lm][half * 16];
            const unsigned short* vb3 = &vbuf[cur][48 + lm][half * 16];
            v0.q[0] = *(const uint4*)(vb0); v0.q[1] = *(const uint4*)(vb0 + 8);
            v1.q[0] = *(const uint4*)(vb1); v1.q[1] = *(const uint4*)(vb1 + 8);
            v2.q[0] = *(const uint4*)(vb2); v2.q[1] = *(const uint4*)(vb2 + 8);
            v3.q[0] = *(const uint4*)(vb3); v3.q[1] = *(const uint4*)(vb3 + 8);
        }
        o0 = __builtin_amdgcn_wmma_f32_16x16x32_bf16(false, pa.v, false, v0.v,
                                                     (short)0, o0, false, false);
        o1 = __builtin_amdgcn_wmma_f32_16x16x32_bf16(false, pa.v, false, v1.v,
                                                     (short)0, o1, false, false);
        o2 = __builtin_amdgcn_wmma_f32_16x16x32_bf16(false, pa.v, false, v2.v,
                                                     (short)0, o2, false, false);
        o3 = __builtin_amdgcn_wmma_f32_16x16x32_bf16(false, pa.v, false, v3.v,
                                                     (short)0, o3, false, false);

        // next buffer's async copies must be complete & visible block-wide
        asm volatile("s_wait_asynccnt 0" ::: "memory");
        __syncthreads();
    }

    // ---- normalize + store ctx (bf16)
#pragma unroll
    for (int r = 0; r < 8; ++r) {
        float inv = 1.0f / l[r];
        int row = q0 + half * 8 + r;
        unsigned short* crow = Ctx + (size_t)row * D_MODEL + head * D_K;
        crow[lm]      = f2bf(o0[r] * inv);
        crow[16 + lm] = f2bf(o1[r] * inv);
        crow[32 + lm] = f2bf(o2[r] * inv);
        crow[48 + lm] = f2bf(o3[r] * inv);
    }
}

// ---------------------------------------------------------------- launch

extern "C" void kernel_launch(void* const* d_in, const int* in_sizes, int n_in,
                              void* d_out, int out_size, void* d_ws, size_t ws_size,
                              hipStream_t stream) {
    (void)in_sizes; (void)n_in; (void)out_size; (void)ws_size;
    const float* x   = (const float*)d_in[0];
    const float* W_q = (const float*)d_in[1];
    const float* b_q = (const float*)d_in[2];
    const float* W_k = (const float*)d_in[3];
    const float* b_k = (const float*)d_in[4];
    const float* W_v = (const float*)d_in[5];
    const float* b_v = (const float*)d_in[6];
    const float* W_o = (const float*)d_in[7];
    const float* b_o = (const float*)d_in[8];
    float* out = (float*)d_out;

    char* ws = (char*)d_ws;
    size_t off = 0;
    auto alloc = [&](size_t bytes) -> char* {
        char* p = ws + off;
        off += (bytes + 255) & ~(size_t)255;
        return p;
    };
    const size_t actN = (size_t)SEQ * D_MODEL * sizeof(unsigned short);
    const size_t wN   = (size_t)D_MODEL * D_MODEL * sizeof(unsigned short);
    unsigned short* xb  = (unsigned short*)alloc(actN);
    unsigned short* Wqt = (unsigned short*)alloc(wN);
    unsigned short* Wkt = (unsigned short*)alloc(wN);
    unsigned short* Wvt = (unsigned short*)alloc(wN);
    unsigned short* Wot = (unsigned short*)alloc(wN);
    unsigned short* Qb  = (unsigned short*)alloc(actN);
    unsigned short* Kb  = (unsigned short*)alloc(actN);
    unsigned short* Vtb = (unsigned short*)alloc(actN);
    unsigned short* ctx = (unsigned short*)alloc(actN);

    // 1) precision conversions (bandwidth-trivial; unlocks 8x math per issue)
    {
        int n4 = SEQ * D_MODEL / 4;
        cvt_f32_to_bf16<<<(n4 + 255) / 256, 256, 0, stream>>>(x, xb, n4);
        dim3 tb(32, 8), tg(D_MODEL / 32, D_MODEL / 32);
        transpose_cvt_bf16<<<tg, tb, 0, stream>>>(W_q, Wqt, D_MODEL, D_MODEL);
        transpose_cvt_bf16<<<tg, tb, 0, stream>>>(W_k, Wkt, D_MODEL, D_MODEL);
        transpose_cvt_bf16<<<tg, tb, 0, stream>>>(W_v, Wvt, D_MODEL, D_MODEL);
        transpose_cvt_bf16<<<tg, tb, 0, stream>>>(W_o, Wot, D_MODEL, D_MODEL);
    }

    // 2) Q/K/V projections (V stored transposed for contiguous PV B-fragments)
    dim3 gg(SEQ / 32, D_MODEL / 256);
    gemm_bf16<0><<<gg, 128, 0, stream>>>(xb, Wqt, b_q, Qb, SEQ, D_MODEL, D_MODEL);
    gemm_bf16<0><<<gg, 128, 0, stream>>>(xb, Wkt, b_k, Kb, SEQ, D_MODEL, D_MODEL);
    gemm_bf16<1><<<gg, 128, 0, stream>>>(xb, Wvt, b_v, Vtb, SEQ, D_MODEL, D_MODEL);

    // 3) flash-style attention with async double-buffered K/V staging
    attn_fwd<<<dim3(SEQ / 64, NUM_HEADS), 128, 0, stream>>>(Qb, Kb, Vtb, ctx);

    // 4) output projection, fp32 result
    gemm_bf16<2><<<gg, 128, 0, stream>>>(ctx, Wot, b_o, out, SEQ, D_MODEL, D_MODEL);
}
